// GPT2Attention_23596550324520
// MI455X (gfx1250) — compile-verified, matrix operand order checked by reading, not run on hardware
//
#include <hip/hip_runtime.h>
#include <hip/hip_bf16.h>

typedef __bf16 bf16_t;
typedef __attribute__((ext_vector_type(16))) __bf16 v16bf;
typedef __attribute__((ext_vector_type(8)))  __bf16 v8bf;
typedef __attribute__((ext_vector_type(8)))  float  v8f;

#define NH     12
#define HD     64
#define HID    768
#define BATCH  2
#define SEQ    2048
#define MROWS  (BATCH*SEQ)   // 4096
#define QKV_N  (3*HID)       // 2304

// Fetch one A/B operand slice for v_wmma_*_16x16x32_bf16 from a row-major row.
// Per ISA 16-bit A layout: lanes 0-15 hold K={k0..k0+7, k0+16..k0+23},
// lanes 16-31 hold K={k0+8..k0+15, k0+24..k0+31}.  Two 16B loads.
static __device__ __forceinline__ v16bf load_ab16(const bf16_t* p, int k0, int hl) {
  const v8bf a = *reinterpret_cast<const v8bf*>(p + k0 + hl * 8);
  const v8bf b = *reinterpret_cast<const v8bf*>(p + k0 + 16 + hl * 8);
  v16bf r;
#pragma unroll
  for (int i = 0; i < 8; ++i) { r[i] = a[i]; r[i + 8] = b[i]; }
  return r;
}

static __device__ __forceinline__ v8f wmma_bf16(v16bf a, v16bf b, v8f c) {
  return __builtin_amdgcn_wmma_f32_16x16x32_bf16(
      /*neg_a=*/false, a, /*neg_b=*/false, b,
      /*c_mod=*/(short)0, c, /*reuse_a=*/false, /*reuse_b=*/false);
}

// ---------------------------------------------------------------- convert
__global__ void cvt_f32_bf16(const float* __restrict__ in, bf16_t* __restrict__ out, int n) {
  int i = blockIdx.x * blockDim.x + threadIdx.x;
  const int stride = gridDim.x * blockDim.x;
  for (; i < n; i += stride) out[i] = (bf16_t)in[i];
}

// ---------------------------------------------------------------- QKV GEMM
// qkv = x @ w_qkvT + b.  32x64 tile per wave: 8 WMMAs per 6 B128-load pairs.
// Scatter Q,K as [b,h,s,d] (bf16), V transposed [b,h,d,s].
__global__ __launch_bounds__(256) void qkv_gemm(
    const bf16_t* __restrict__ xb, const bf16_t* __restrict__ wb,
    const float* __restrict__ bias,
    bf16_t* __restrict__ Qo, bf16_t* __restrict__ Ko, bf16_t* __restrict__ Vt) {
  const int lane = threadIdx.x & 31;
  const int wid  = threadIdx.x >> 5;
  const int hl   = lane >> 4;
  const int lo   = lane & 15;
  const int tile = blockIdx.x * (blockDim.x >> 5) + wid;
  const int TN   = QKV_N / 64;                 // 36 tiles in N
  const int tm   = tile / TN;                  // 0..127 (rows of 32)
  const int tn   = tile % TN;
  if (tm >= MROWS / 32) return;                // wave-uniform

  const bf16_t* arow0 = xb + (size_t)(tm * 32 + lo) * HID;
  const bf16_t* arow1 = arow0 + (size_t)16 * HID;
  const bf16_t* brow[4];
#pragma unroll
  for (int ni = 0; ni < 4; ++ni)
    brow[ni] = wb + (size_t)(tn * 64 + ni * 16 + lo) * HID;

  v8f acc[2][4] = {};
#pragma unroll 2
  for (int k0 = 0; k0 < HID; k0 += 32) {
    const v16bf a0 = load_ab16(arow0, k0, hl);
    const v16bf a1 = load_ab16(arow1, k0, hl);
#pragma unroll
    for (int ni = 0; ni < 4; ++ni) {
      const v16bf b = load_ab16(brow[ni], k0, hl);
      acc[0][ni] = wmma_bf16(a0, b, acc[0][ni]);
      acc[1][ni] = wmma_bf16(a1, b, acc[1][ni]);
    }
  }

#pragma unroll
  for (int ni = 0; ni < 4; ++ni) {
    const int   ncol = tn * 64 + ni * 16 + lo;
    const float bv   = bias[ncol];
#pragma unroll
    for (int mi = 0; mi < 2; ++mi) {
#pragma unroll
      for (int r = 0; r < 8; ++r) {
        const int   m  = tm * 32 + mi * 16 + r + 8 * hl;  // C layout row
        const int   bb = m >> 11;                         // / SEQ
        const int   s  = m & (SEQ - 1);
        const float v  = acc[mi][ni][r] + bv;
        if (ncol < HID) {
          const int h = ncol >> 6, d = ncol & 63;
          Qo[((size_t)(bb * NH + h) * SEQ + s) * HD + d] = (bf16_t)v;
        } else if (ncol < 2 * HID) {
          const int nc = ncol - HID, h = nc >> 6, d = nc & 63;
          Ko[((size_t)(bb * NH + h) * SEQ + s) * HD + d] = (bf16_t)v;
        } else {
          const int nc = ncol - 2 * HID, h = nc >> 6, d = nc & 63;
          Vt[((size_t)(bb * NH + h) * HD + d) * SEQ + s] = (bf16_t)v;  // V^T
        }
      }
    }
  }
}

// ---------------------------------------------------------------- attention
// One wave per (b, h, 16-query tile); streaming causal softmax (flash style).
__global__ __launch_bounds__(256) void attn_kernel(
    const bf16_t* __restrict__ Qo, const bf16_t* __restrict__ Ko,
    const bf16_t* __restrict__ Vt, bf16_t* __restrict__ attnb) {
  __shared__ bf16_t lds_p[8][16][32];          // per-wave P staging (C->A layout hop)
  const int lane = threadIdx.x & 31;
  const int wid  = threadIdx.x >> 5;
  const int hl   = lane >> 4;
  const int lo   = lane & 15;
  const int gid  = blockIdx.x * (blockDim.x >> 5) + wid;
  const int QT   = SEQ / 16;                   // 128
  const int bb   = gid / (NH * QT);
  const int rem  = gid % (NH * QT);
  const int h    = rem / QT;
  const int q0   = (rem % QT) * 16;

  const bf16_t* Qbase = Qo + (size_t)(bb * NH + h) * SEQ * HD;
  const bf16_t* Kbase = Ko + (size_t)(bb * NH + h) * SEQ * HD;
  const bf16_t* Vbase = Vt + (size_t)(bb * NH + h) * HD * SEQ;

  // Preload Q A-operands for both 32-wide K chunks of hd=64 (A row = lane&15).
  const bf16_t* qrow = Qbase + (size_t)(q0 + lo) * HD;
  const v16bf aq0 = load_ab16(qrow, 0, hl);
  const v16bf aq1 = load_ab16(qrow, 32, hl);

  v8f   O[4] = {};                             // 16q x 64d accumulator, C layout
  float mi[8], li[8];
#pragma unroll
  for (int r = 0; r < 8; ++r) { mi[r] = -__builtin_inff(); li[r] = 0.f; }

  for (int j = 0; j < q0 + 16; j += 32) {      // 32 keys per step
    // scores: two 16x16 tiles (keys j..j+15 and j+16..j+31)
    const bf16_t* krow0 = Kbase + (size_t)(j + lo) * HD;
    v8f c0 = {};
    c0 = wmma_bf16(aq0, load_ab16(krow0, 0, hl), c0);
    c0 = wmma_bf16(aq1, load_ab16(krow0, 32, hl), c0);
    const bf16_t* krow1 = Kbase + (size_t)(j + 16 + lo) * HD;
    v8f c1 = {};
    c1 = wmma_bf16(aq0, load_ab16(krow1, 0, hl), c1);
    c1 = wmma_bf16(aq1, load_ab16(krow1, 32, hl), c1);

    float p0[8], p1[8];
#pragma unroll
    for (int r = 0; r < 8; ++r) {
      const int rowg = q0 + r + 8 * hl;
      float s0 = c0[r] * 0.125f;               // 1/sqrt(64)
      float s1 = c1[r] * 0.125f;
      if (j + lo > rowg)      s0 = -__builtin_inff();   // causal mask
      if (j + 16 + lo > rowg) s1 = -__builtin_inff();
      float rm = fmaxf(s0, s1);                // row max over 32 keys (16 lanes/half)
      rm = fmaxf(rm, __shfl_xor(rm, 1, 32));
      rm = fmaxf(rm, __shfl_xor(rm, 2, 32));
      rm = fmaxf(rm, __shfl_xor(rm, 4, 32));
      rm = fmaxf(rm, __shfl_xor(rm, 8, 32));
      const float mnew  = fmaxf(mi[r], rm);
      const float alpha = __expf(mi[r] - mnew);
      mi[r] = mnew;
      p0[r] = __expf(s0 - mnew);
      p1[r] = __expf(s1 - mnew);
      float rs = p0[r] + p1[r];
      rs += __shfl_xor(rs, 1, 32);
      rs += __shfl_xor(rs, 2, 32);
      rs += __shfl_xor(rs, 4, 32);
      rs += __shfl_xor(rs, 8, 32);
      li[r] = li[r] * alpha + rs;
#pragma unroll
      for (int t = 0; t < 4; ++t) O[t][r] *= alpha;
    }

    // P: C layout -> LDS (row-major 16x32) -> A-operand layout. Intra-wave
    // only, so a split-counter wait is enough (DS ops are in-order per wave).
#pragma unroll
    for (int r = 0; r < 8; ++r) {
      lds_p[wid][r + 8 * hl][lo]      = (bf16_t)p0[r];
      lds_p[wid][r + 8 * hl][16 + lo] = (bf16_t)p1[r];
    }
    asm volatile("s_wait_dscnt 0x0" ::: "memory");
    const v16bf pa = load_ab16(&lds_p[wid][lo][0], 0, hl);

    // O += P @ V ; B operand = rows of V^T (contiguous along s)
#pragma unroll
    for (int t = 0; t < 4; ++t) {
      const bf16_t* vrow = Vbase + (size_t)(t * 16 + lo) * SEQ;
      O[t] = wmma_bf16(pa, load_ab16(vrow, j, hl), O[t]);
    }
  }

#pragma unroll
  for (int t = 0; t < 4; ++t)
#pragma unroll
    for (int r = 0; r < 8; ++r) {
      const int rowg = q0 + r + 8 * hl;
      attnb[((size_t)(bb * SEQ + rowg)) * HID + h * HD + t * 16 + lo] =
          (bf16_t)(O[t][r] / li[r]);
    }
}

// ---------------------------------------------------------------- out proj
// out = attn @ w_oT + b.  32x64 tile per wave, fp32 output.
__global__ __launch_bounds__(256) void oproj_gemm(
    const bf16_t* __restrict__ ab, const bf16_t* __restrict__ wb,
    const float* __restrict__ bias, float* __restrict__ out) {
  const int lane = threadIdx.x & 31;
  const int wid  = threadIdx.x >> 5;
  const int hl   = lane >> 4;
  const int lo   = lane & 15;
  const int tile = blockIdx.x * (blockDim.x >> 5) + wid;
  const int TN   = HID / 64;                   // 12
  const int tm   = tile / TN;
  const int tn   = tile % TN;
  if (tm >= MROWS / 32) return;

  const bf16_t* arow0 = ab + (size_t)(tm * 32 + lo) * HID;
  const bf16_t* arow1 = arow0 + (size_t)16 * HID;
  const bf16_t* brow[4];
#pragma unroll
  for (int ni = 0; ni < 4; ++ni)
    brow[ni] = wb + (size_t)(tn * 64 + ni * 16 + lo) * HID;

  v8f acc[2][4] = {};
#pragma unroll 2
  for (int k0 = 0; k0 < HID; k0 += 32) {
    const v16bf a0 = load_ab16(arow0, k0, hl);
    const v16bf a1 = load_ab16(arow1, k0, hl);
#pragma unroll
    for (int ni = 0; ni < 4; ++ni) {
      const v16bf b = load_ab16(brow[ni], k0, hl);
      acc[0][ni] = wmma_bf16(a0, b, acc[0][ni]);
      acc[1][ni] = wmma_bf16(a1, b, acc[1][ni]);
    }
  }

#pragma unroll
  for (int ni = 0; ni < 4; ++ni) {
    const int   n  = tn * 64 + ni * 16 + lo;
    const float bv = bias[n];
#pragma unroll
    for (int mi = 0; mi < 2; ++mi)
#pragma unroll
      for (int r = 0; r < 8; ++r)
        out[(size_t)(tm * 32 + mi * 16 + r + 8 * hl) * HID + n] = acc[mi][ni][r] + bv;
  }
}

// ---------------------------------------------------------------- launch
extern "C" void kernel_launch(void* const* d_in, const int* in_sizes, int n_in,
                              void* d_out, int out_size, void* d_ws, size_t ws_size,
                              hipStream_t stream) {
  const float* x     = (const float*)d_in[0];
  const float* w_qkv = (const float*)d_in[1];
  const float* b_qkv = (const float*)d_in[2];
  const float* w_o   = (const float*)d_in[3];
  const float* b_o   = (const float*)d_in[4];
  float*       out   = (float*)d_out;
  (void)in_sizes; (void)n_in; (void)out_size; (void)ws_size;

  char* w = (char*)d_ws;
  auto take = [&](size_t bytes) {
    char* p = w;
    w += (bytes + 255) & ~(size_t)255;
    return p;
  };
  bf16_t* xb    = (bf16_t*)take((size_t)MROWS * HID * 2);
  bf16_t* wqkvb = (bf16_t*)take((size_t)QKV_N * HID * 2);
  bf16_t* wob   = (bf16_t*)take((size_t)HID * HID * 2);
  bf16_t* Qo    = (bf16_t*)take((size_t)MROWS * HID * 2);
  bf16_t* Ko    = (bf16_t*)take((size_t)MROWS * HID * 2);
  bf16_t* Vt    = (bf16_t*)take((size_t)MROWS * HID * 2);
  bf16_t* attnb = (bf16_t*)take((size_t)MROWS * HID * 2);

  cvt_f32_bf16<<<2048, 256, 0, stream>>>(x, xb, MROWS * HID);
  cvt_f32_bf16<<<2048, 256, 0, stream>>>(w_qkv, wqkvb, QKV_N * HID);
  cvt_f32_bf16<<<1024, 256, 0, stream>>>(w_o, wob, HID * HID);

  qkv_gemm<<<(MROWS / 32) * (QKV_N / 64) / 8, 256, 0, stream>>>(xb, wqkvb, b_qkv, Qo, Ko, Vt);
  attn_kernel<<<(BATCH * NH * (SEQ / 16)) / 8, 256, 0, stream>>>(Qo, Ko, Vt, attnb);
  oproj_gemm<<<(MROWS / 32) * (HID / 64) / 8, 256, 0, stream>>>(attnb, wob, b_o, out);
}
